// ProbSparseAttention_36361193128144
// MI455X (gfx1250) — matrix operand order repaired and checked by reading, not compiled, not run on hardware
//
#include <hip/hip_runtime.h>
#include <hip/hip_bf16.h>

// ProbSparse attention for MI455X (gfx1250), wave32 + WMMA f16->f32.
// B=4, L=4096, H=16, D=64, n_top = int(4096*0.02*ln(4096)) = 681.

typedef __attribute__((ext_vector_type(16))) _Float16 v16h;
typedef __attribute__((ext_vector_type(8)))  float    v8f;

#define BB   4
#define LL   4096
#define HH   16
#define DD   64
#define NTOP 681
#define NFULL 672                     // 42 full 16-column chunks
#define QT_TILES   43                 // ceil(681/16)
#define UPAD       (QT_TILES * 16)    // 688 sampled rows, zero padded
#define SCALE 0.125f                  // 1/sqrt(64)
#define ROWSTRIDE ((size_t)HH * DD)   // stride between sequence positions (floats)

#ifndef __has_builtin
#define __has_builtin(x) 0
#endif
#if __has_builtin(__builtin_amdgcn_global_load_async_to_lds_b128) && \
    __has_builtin(__builtin_amdgcn_s_wait_asynccnt)
#define USE_ASYNC_LDS 1
// ROCm signature (from diagnostics): param1 = int4 __device__* (global src),
// param2 = LDS dst, params 3/4 = imm offset / imm cpol.
typedef int v4i_g __attribute__((vector_size(16)));
typedef __attribute__((address_space(1))) v4i_g* gv4i_p;
typedef __attribute__((address_space(3))) v4i_g* sv4i_p;
#else
#define USE_ASYNC_LDS 0
#endif

__device__ __forceinline__ v8f wmma_f16(v16h a, v16h b, v8f c) {
  return __builtin_amdgcn_wmma_f32_16x16x32_f16(false, a, false, b, (short)0, c,
                                                false, false);
}

// A-operand (16x32 f16): lane&15 = row; half h = lane>>4 selects K range
// [16h,16h+16).
__device__ __forceinline__ v16h load_A_row_f32(const float* row, int h, float scale) {
  v16h a;
  const int k0 = 16 * h, k1 = 16 * h + 8;
#pragma unroll
  for (int i = 0; i < 8; ++i) {
    a[i]     = (_Float16)(row[k0 + i] * scale);
    a[8 + i] = (_Float16)(row[k1 + i] * scale);
  }
  return a;
}

__device__ __forceinline__ v16h load_A_row_h(const _Float16* row, int h) {
  v16h a;
  const int k0 = 16 * h, k1 = 16 * h + 8;
#pragma unroll
  for (int i = 0; i < 8; ++i) {
    a[i]     = row[k0 + i];
    a[8 + i] = row[k1 + i];
  }
  return a;
}

// B-operand (32x16 f16): lane&15 = column N, half selects 16 consecutive K.
__device__ __forceinline__ v16h load_B_contig(const _Float16* p) {
  v16h b;
#pragma unroll
  for (int i = 0; i < 16; ++i) b[i] = p[i];
  return b;
}

// ---------------------------------------------------------------------------
// Kernel 0: zero the output (non-selected rows must be 0).
// ---------------------------------------------------------------------------
__global__ void zero_out_kernel(float* __restrict__ out, int n) {
  int i = blockIdx.x * blockDim.x + threadIdx.x;
  int stride = gridDim.x * blockDim.x;
  for (; i < n; i += stride) out[i] = 0.0f;
}

// ---------------------------------------------------------------------------
// Kernel 1: gather + convert the 681 sampled K rows per (b,h) to f16 once.
// Ks16[bh][u][d], padded with zeros to UPAD rows.
// ---------------------------------------------------------------------------
__global__ __launch_bounds__(256) void pack_samples_kernel(const float* __restrict__ K,
                                                           _Float16* __restrict__ Ks16) {
  const int blk = blockIdx.x;                 // B*H*QT_TILES blocks
  const int ut  = blk % QT_TILES;
  const int bh  = blk / QT_TILES;
  const int b   = bh / HH, h = bh % HH;
  const float* Kbh = K + ((size_t)b * LL * HH + h) * DD;
  _Float16* dst = Ks16 + ((size_t)bh * UPAD + ut * 16) * DD;
  for (int t = threadIdx.x; t < 16 * DD; t += 256) {
    const int r = t >> 6, d = t & 63;
    const int u = ut * 16 + r;
    float v = 0.0f;
    if (u < NTOP) {
      const int kidx = (u * 1997 + 71) & (LL - 1);  // pseudo-permutation sample
      v = Kbh[(size_t)kidx * ROWSTRIDE + d];
    }
    dst[t] = (_Float16)v;
  }
}

// ---------------------------------------------------------------------------
// Kernel 2: sparsity scoring.  M[b,h,q] = max_u(Q.Ks^T) - sum_u(...)/L.
// 4 waves per block share the staged sample tile; each wave owns a q-tile.
// Tail chunk is peeled so the 42 hot iterations carry no column mask.
// ---------------------------------------------------------------------------
__global__ __launch_bounds__(128) void score_kernel(const float* __restrict__ Q,
                                                    const _Float16* __restrict__ Ks16,
                                                    float* __restrict__ M) {
  const int blk  = blockIdx.x;                 // B*H*64 blocks
  const int qg   = blk % (LL / 64);            // group of 4 q-tiles
  const int bh   = blk / (LL / 64);
  const int b    = bh / HH, h = bh % HH;
  const int tid  = threadIdx.x;
  const int lane = tid & 31, wv = tid >> 5;
  const int rown = lane & 15, half = lane >> 4, kb = half * 16;

  __shared__ _Float16 ldsKs[16 * DD];          // one 16-sample chunk, shared

  const int qtile = qg * 4 + wv;               // 0..255
  const int qrow  = qtile * 16 + rown;
  const float* qptr = Q + (((size_t)b * LL + qrow) * HH + h) * DD;
  const v16h a0 = load_A_row_f32(qptr,      half, 1.0f);
  const v16h a1 = load_A_row_f32(qptr + 32, half, 1.0f);

  const _Float16* Ksbh = Ks16 + (size_t)bh * UPAD * DD;

  float rmax[8], rsum[8];
#pragma unroll
  for (int j = 0; j < 8; ++j) { rmax[j] = -3.0e38f; rsum[j] = 0.0f; }

  // ---- 42 full chunks, no column masking ----
  for (int u0 = 0; u0 < NFULL; u0 += 16) {
    ((uint4*)ldsKs)[tid] = ((const uint4*)(Ksbh + (size_t)u0 * DD))[tid];
    __syncthreads();

    const v16h b0 = load_B_contig(&ldsKs[rown * DD + kb]);        // d [0,32)
    const v16h b1 = load_B_contig(&ldsKs[rown * DD + 32 + kb]);   // d [32,64)
    v8f c = {};
    c = wmma_f16(a0, b0, c);
    c = wmma_f16(a1, b1, c);
#pragma unroll
    for (int j = 0; j < 8; ++j) {
      rmax[j] = fmaxf(rmax[j], c[j]);
      rsum[j] += c[j];
    }
    __syncthreads();
  }

  // ---- peeled tail chunk (columns NFULL..NTOP-1 valid) ----
  {
    ((uint4*)ldsKs)[tid] = ((const uint4*)(Ksbh + (size_t)NFULL * DD))[tid];
    __syncthreads();
    const v16h b0 = load_B_contig(&ldsKs[rown * DD + kb]);
    const v16h b1 = load_B_contig(&ldsKs[rown * DD + 32 + kb]);
    v8f c = {};
    c = wmma_f16(a0, b0, c);
    c = wmma_f16(a1, b1, c);
    const bool valid = (NFULL + rown) < NTOP;
#pragma unroll
    for (int j = 0; j < 8; ++j) {
      const float s = c[j];
      if (valid) { rmax[j] = fmaxf(rmax[j], s); rsum[j] += s; }
    }
  }

#pragma unroll
  for (int j = 0; j < 8; ++j) {
#pragma unroll
    for (int mk = 1; mk < 16; mk <<= 1) {
      rmax[j] = fmaxf(rmax[j], __shfl_xor(rmax[j], mk, 32));
      rsum[j] += __shfl_xor(rsum[j], mk, 32);
    }
  }
  if (rown == 0) {
    const int rowbase = half * 8;              // C layout rows j / j+8
#pragma unroll
    for (int j = 0; j < 8; ++j) {
      M[(size_t)bh * LL + qtile * 16 + rowbase + j] =
          rmax[j] - rsum[j] * (1.0f / (float)LL);
    }
  }
}

// ---------------------------------------------------------------------------
// Kernel 3: per-(b,h) top-681 selection via 8-bit radix select + compaction.
// ---------------------------------------------------------------------------
__global__ __launch_bounds__(256) void topk_kernel(const float* __restrict__ M,
                                                   int* __restrict__ sel) {
  const int bh  = blockIdx.x;
  const int tid = threadIdx.x;

  __shared__ unsigned skey[LL];
  __shared__ int hist[256];
  __shared__ unsigned s_prefix;
  __shared__ int s_remaining, s_cnt;

  const float* Mp = M + (size_t)bh * LL;
  for (int i = tid; i < LL; i += 256) {
    unsigned u = __float_as_uint(Mp[i]);
    u = (u & 0x80000000u) ? ~u : (u | 0x80000000u);   // order-preserving key
    skey[i] = u;
  }
  if (tid == 0) { s_prefix = 0u; s_remaining = NTOP; }
  __syncthreads();

  for (int shift = 24; shift >= 0; shift -= 8) {
    hist[tid] = 0;
    __syncthreads();
    const unsigned pm   = (shift == 24) ? 0u : (0xFFFFFFFFu << (shift + 8));
    const unsigned pref = s_prefix;
    for (int i = tid; i < LL; i += 256)
      if ((skey[i] & pm) == pref) atomicAdd(&hist[(skey[i] >> shift) & 255], 1);
    __syncthreads();
    if (tid == 0) {
      int rem = s_remaining, bsel = 0;
      for (int bb = 255; bb >= 0; --bb) {
        if (rem > hist[bb]) rem -= hist[bb];
        else { bsel = bb; break; }
      }
      s_prefix    = pref | ((unsigned)bsel << shift);
      s_remaining = rem;
    }
    __syncthreads();
  }

  const unsigned T = s_prefix;   // key of the NTOP-th largest element
  if (tid == 0) s_cnt = 0;
  __syncthreads();
  int* selp = sel + (size_t)bh * NTOP;
  for (int i = tid; i < LL; i += 256)
    if (skey[i] > T) { int p = atomicAdd(&s_cnt, 1); if (p < NTOP) selp[p] = i; }
  __syncthreads();
  for (int i = tid; i < LL; i += 256)
    if (skey[i] == T) { int p = atomicAdd(&s_cnt, 1); if (p < NTOP) selp[p] = i; }
}

// ---------------------------------------------------------------------------
// Kernel 4: flash attention over selected queries, scattered output.
// 4 waves/block share staged K/V tiles; each wave = 16 selected queries.
// K tile staged via GLOBAL_LOAD_ASYNC_TO_LDS when the toolchain exposes it.
// ---------------------------------------------------------------------------
#define BLKS_PER_BH 11   // ceil(43/4) blocks of 4 q-tiles

__global__ __launch_bounds__(128) void attn_kernel(const float* __restrict__ Q,
                                                   const float* __restrict__ K,
                                                   const float* __restrict__ V,
                                                   const int* __restrict__ sel,
                                                   float* __restrict__ out) {
  const int blk  = blockIdx.x;
  const int tg   = blk % BLKS_PER_BH;
  const int bh   = blk / BLKS_PER_BH;
  const int b    = bh / HH, h = bh % HH;
  const int tid  = threadIdx.x;
  const int lane = tid & 31, wv = tid >> 5;
  const int rown = lane & 15, half = lane >> 4, kb = half * 16;
  const int qt   = tg * 4 + wv;                // 0..43 (44th tile is inert)

  __shared__ _Float16 ldsK[32 * DD];           // K tile f16, row-major [key][d]
  __shared__ _Float16 ldsVT[DD * 32];          // V tile f16, transposed [d][key]
  __shared__ _Float16 ldsP[4][16 * 32];        // per-wave softmax probs
#if USE_ASYNC_LDS
  __shared__ float ldsKf[32 * DD];             // async-landed f32 K tile
#endif

  int selpos = qt * 16 + rown;
  if (selpos >= NTOP) selpos = NTOP - 1;
  const int q = sel[(size_t)bh * NTOP + selpos];
  const float* qptr = Q + (((size_t)b * LL + q) * HH + h) * DD;
  const v16h a0 = load_A_row_f32(qptr,      half, SCALE);
  const v16h a1 = load_A_row_f32(qptr + 32, half, SCALE);

  const float* Kbh = K + ((size_t)b * LL * HH + h) * DD;
  const float* Vbh = V + ((size_t)b * LL * HH + h) * DD;

  float m[8], l[8];
  v8f O0 = {}, O1 = {}, O2 = {}, O3 = {};
#pragma unroll
  for (int j = 0; j < 8; ++j) { m[j] = -3.0e38f; l[j] = 0.0f; }

  for (int k0 = 0; k0 < LL; k0 += 32) {
#if USE_ASYNC_LDS
    // ---- K tile: direct global->LDS async b128 transfers (ASYNCcnt path) ----
#pragma unroll
    for (int i = 0; i < 4; ++i) {
      const int idx = tid + i * 128;           // 0..511 16B slots
      const int r   = idx >> 4;
      const int c4  = (idx & 15) * 4;          // float base within row
      __builtin_amdgcn_global_load_async_to_lds_b128(
          (gv4i_p)(void*)(Kbh + (size_t)(k0 + r) * ROWSTRIDE + c4),
          (sv4i_p)&ldsKf[r * DD + c4],
          0, 0);
    }
#endif
    // ---- V tile: load f32, convert, store transposed f16 ----
#pragma unroll
    for (int i = 0; i < 4; ++i) {
      const int idx = tid + i * 128;           // 0..511 float4 slots
      const int r   = idx >> 4;                // key row 0..31
      const int d4  = (idx & 15) * 4;          // d base
      const float4 v4 = ((const float4*)(Vbh + (size_t)(k0 + r) * ROWSTRIDE))[idx & 15];
      ldsVT[(d4 + 0) * 32 + r] = (_Float16)v4.x;
      ldsVT[(d4 + 1) * 32 + r] = (_Float16)v4.y;
      ldsVT[(d4 + 2) * 32 + r] = (_Float16)v4.z;
      ldsVT[(d4 + 3) * 32 + r] = (_Float16)v4.w;
#if !USE_ASYNC_LDS
      const float4 k4 = ((const float4*)(Kbh + (size_t)(k0 + r) * ROWSTRIDE))[idx & 15];
      union { _Float16 hx[4]; uint2 u; } pk;
      pk.hx[0] = (_Float16)k4.x; pk.hx[1] = (_Float16)k4.y;
      pk.hx[2] = (_Float16)k4.z; pk.hx[3] = (_Float16)k4.w;
      *(uint2*)&ldsK[r * DD + d4] = pk.u;      // 8B aligned LDS store
#endif
    }
#if USE_ASYNC_LDS
    __builtin_amdgcn_s_wait_asynccnt(0);
    __syncthreads();
    // convert the async-landed f32 K tile to f16 operand tile
#pragma unroll
    for (int i = 0; i < 4; ++i) {
      const int idx = tid + i * 128;
      const float4 k4 = ((const float4*)ldsKf)[idx];
      union { _Float16 hx[4]; uint2 u; } pk;
      pk.hx[0] = (_Float16)k4.x; pk.hx[1] = (_Float16)k4.y;
      pk.hx[2] = (_Float16)k4.z; pk.hx[3] = (_Float16)k4.w;
      ((uint2*)ldsK)[idx] = pk.u;
    }
#endif
    __syncthreads();

    if (tid < 32 && k0 + 32 < LL) {            // hint next chunk to caches
      __builtin_prefetch(&Kbh[(size_t)(k0 + 32 + tid) * ROWSTRIDE], 0, 0);
      __builtin_prefetch(&Vbh[(size_t)(k0 + 32 + tid) * ROWSTRIDE], 0, 0);
    }

    // ---- S = Q_sel . K^T : two 16x16 column tiles, K-dim 64 = 2 WMMAs each
    v8f S0 = {}, S1 = {};
    {
      const v16h bA = load_B_contig(&ldsK[rown * DD + kb]);
      const v16h bB = load_B_contig(&ldsK[rown * DD + 32 + kb]);
      S0 = wmma_f16(a0, bA, S0);
      S0 = wmma_f16(a1, bB, S0);
      const v16h bC = load_B_contig(&ldsK[(16 + rown) * DD + kb]);
      const v16h bD = load_B_contig(&ldsK[(16 + rown) * DD + 32 + kb]);
      S1 = wmma_f16(a0, bC, S1);
      S1 = wmma_f16(a1, bD, S1);
    }

    // ---- online softmax (rows of this lane's half) ----
    float cmax[8];
#pragma unroll
    for (int j = 0; j < 8; ++j) cmax[j] = fmaxf(S0[j], S1[j]);
#pragma unroll
    for (int j = 0; j < 8; ++j)
#pragma unroll
      for (int mk = 1; mk < 16; mk <<= 1)
        cmax[j] = fmaxf(cmax[j], __shfl_xor(cmax[j], mk, 32));

    float alpha[8], csum[8], p0[8], p1[8];
#pragma unroll
    for (int j = 0; j < 8; ++j) {
      const float mn = fmaxf(m[j], cmax[j]);
      alpha[j] = __expf(m[j] - mn);
      m[j] = mn;
      p0[j] = __expf(S0[j] - mn);
      p1[j] = __expf(S1[j] - mn);
      csum[j] = p0[j] + p1[j];
    }
#pragma unroll
    for (int j = 0; j < 8; ++j)
#pragma unroll
      for (int mk = 1; mk < 16; mk <<= 1)
        csum[j] += __shfl_xor(csum[j], mk, 32);
#pragma unroll
    for (int j = 0; j < 8; ++j) l[j] = l[j] * alpha[j] + csum[j];

#pragma unroll
    for (int j = 0; j < 8; ++j) {
      O0[j] *= alpha[j]; O1[j] *= alpha[j]; O2[j] *= alpha[j]; O3[j] *= alpha[j];
    }

    // ---- route P through per-wave LDS tile: C layout -> A layout ----
    __syncthreads();
#pragma unroll
    for (int j = 0; j < 8; ++j) {
      const int row = j + 8 * half;
      ldsP[wv][row * 32 + rown]      = (_Float16)p0[j];
      ldsP[wv][row * 32 + 16 + rown] = (_Float16)p1[j];
    }
    __syncthreads();

    const v16h aP = load_A_row_h(&ldsP[wv][rown * 32], half);

    // ---- O += P . V : 4 WMMAs, one per 16-wide d-chunk ----
    O0 = wmma_f16(aP, load_B_contig(&ldsVT[(0 * 16 + rown) * 32 + kb]), O0);
    O1 = wmma_f16(aP, load_B_contig(&ldsVT[(1 * 16 + rown) * 32 + kb]), O1);
    O2 = wmma_f16(aP, load_B_contig(&ldsVT[(2 * 16 + rown) * 32 + kb]), O2);
    O3 = wmma_f16(aP, load_B_contig(&ldsVT[(3 * 16 + rown) * 32 + kb]), O3);
    __syncthreads();
  }

  // ---- finalize and scatter: out[b, q_r, h, :] = O_row / l_row ----
#pragma unroll
  for (int j = 0; j < 8; ++j) {
    const int r  = j + 8 * half;               // C-layout row for this lane
    const int sp = qt * 16 + r;
    if (sp < NTOP) {
      const int qi = sel[(size_t)bh * NTOP + sp];
      const float inv = 1.0f / l[j];
      float* op = out + (((size_t)b * LL + qi) * HH + h) * DD;
      op[0 * 16 + rown] = O0[j] * inv;
      op[1 * 16 + rown] = O1[j] * inv;
      op[2 * 16 + rown] = O2[j] * inv;
      op[3 * 16 + rown] = O3[j] * inv;
    }
  }
}

// ---------------------------------------------------------------------------
extern "C" void kernel_launch(void* const* d_in, const int* in_sizes, int n_in,
                              void* d_out, int out_size, void* d_ws, size_t ws_size,
                              hipStream_t stream) {
  const float* Q = (const float*)d_in[0];
  const float* K = (const float*)d_in[1];
  const float* V = (const float*)d_in[2];
  float* out = (float*)d_out;

  // workspace: packed f16 samples | M scores | selected indices
  char* ws = (char*)d_ws;
  size_t ks_bytes = (size_t)BB * HH * UPAD * DD * sizeof(_Float16);   // ~5.6 MB
  ks_bytes = (ks_bytes + 255) & ~(size_t)255;
  size_t m_bytes = (size_t)BB * HH * LL * sizeof(float);              // 1 MB
  m_bytes = (m_bytes + 255) & ~(size_t)255;

  _Float16* Ks16 = (_Float16*)ws;
  float*    Mws  = (float*)(ws + ks_bytes);
  int*      selws = (int*)(ws + ks_bytes + m_bytes);

  const int total = BB * LL * HH * DD;
  zero_out_kernel<<<2048, 256, 0, stream>>>(out, total);

  pack_samples_kernel<<<BB * HH * QT_TILES, 256, 0, stream>>>(K, Ks16);

  score_kernel<<<BB * HH * (LL / 64), 128, 0, stream>>>(Q, Ks16, Mws);

  topk_kernel<<<BB * HH, 256, 0, stream>>>(Mws, selws);

  attn_kernel<<<BB * HH * BLKS_PER_BH, 128, 0, stream>>>(Q, K, V, selws, out);
}
